// HyenaBlock_38878043964091
// MI455X (gfx1250) — matrix operand order repaired and checked by reading, not compiled, not run on hardware
//
#include <hip/hip_runtime.h>
#include <hip/hip_bf16.h>
#include <stdint.h>

// ---------------------------------------------------------------------------
// Hyena block for MI455X (gfx1250, wave32, WMMA + async-to-LDS).
//   B=2, L=8192, D=1024.
//   GEMMs: v_wmma_f32_16x16x32_bf16, 3x-bf16 split (near-fp32 accuracy),
//          operands pre-split to bf16 hi/lo, staged with
//          global_load_async_to_lds_b128 into double-buffered LDS.
//   FFT conv -> truncated causal conv (exp(-0.01 t) -> 1664 effective taps).
// ---------------------------------------------------------------------------

#define BB 2
#define LL 8192
#define DD 1024
#define MM (BB * LL)        // 16384 rows
#define KTRUNC 1664         // decay e^-16.6 ~ 6e-8 -> truncation error ~1e-6

typedef float v4f __attribute__((ext_vector_type(4)));
typedef float v8f __attribute__((ext_vector_type(8)));
typedef __bf16 v16bf __attribute__((ext_vector_type(16)));
typedef unsigned int u32x4 __attribute__((ext_vector_type(4)));
typedef unsigned int u32x2 __attribute__((ext_vector_type(2)));

union Frag {
    v16bf bf;
    u32x4 q[2];
};

__device__ __forceinline__ unsigned short f2bf_rn(float f) {
    unsigned int x = __float_as_uint(f);
    unsigned int r = x + 0x7fffu + ((x >> 16) & 1u);   // round to nearest even
    return (unsigned short)(r >> 16);
}
__device__ __forceinline__ float bf2f(unsigned short h) {
    return __uint_as_float(((unsigned int)h) << 16);
}
__device__ __forceinline__ unsigned lds_addr_of(const void* p) {
    // flat LDS aperture: low 32 bits are the wave-relative LDS byte address
    return (unsigned)(unsigned long long)p;
}
// CDNA5 async copy: 16B per lane, per-lane LDS destination address (ASYNCcnt)
__device__ __forceinline__ void async_b128(unsigned lds, unsigned voff, const void* sbase) {
    asm volatile("global_load_async_to_lds_b128 %0, %1, %2"
                 :: "v"(lds), "v"(voff), "s"(sbase) : "memory");
}
__device__ __forceinline__ void wait_async0() {
    asm volatile("s_wait_asynccnt 0x0" ::: "memory");
}

// ---------------------------------------------------------------------------
// Pre-split fp32 -> bf16 hi + bf16 residual (element-wise, vectorized).
// ---------------------------------------------------------------------------
__global__ __launch_bounds__(256)
void split_bf16_kernel(const float* __restrict__ in, unsigned short* __restrict__ hi,
                       unsigned short* __restrict__ lo, int n4) {
    int i = blockIdx.x * 256 + threadIdx.x;     // float4 index
    if (i >= n4) return;
    v4f f = *(const v4f*)(in + (size_t)i * 4);
    unsigned short h[4], l[4];
#pragma unroll
    for (int e = 0; e < 4; ++e) {
        h[e] = f2bf_rn(f[e]);
        l[e] = f2bf_rn(f[e] - bf2f(h[e]));
    }
    u32x2 H, L;
    H[0] = (unsigned)h[0] | ((unsigned)h[1] << 16);
    H[1] = (unsigned)h[2] | ((unsigned)h[3] << 16);
    L[0] = (unsigned)l[0] | ((unsigned)l[1] << 16);
    L[1] = (unsigned)l[2] | ((unsigned)l[3] << 16);
    *(u32x2*)(hi + (size_t)i * 4) = H;
    *(u32x2*)(lo + (size_t)i * 4) = L;
}

// ---------------------------------------------------------------------------
// Transpose + split for weights: Th[n*K+k] = bf16_hi(W[k*N+n]) etc.
// ---------------------------------------------------------------------------
__global__ __launch_bounds__(256)
void splitT_bf16_kernel(const float* __restrict__ W,
                        unsigned short* __restrict__ Th, unsigned short* __restrict__ Tl,
                        int Kdim, int Ndim) {
    __shared__ float tile[32][33];
    const int n0 = blockIdx.x * 32;
    const int k0 = blockIdx.y * 32;
    const int tid = threadIdx.x;
    const int r  = tid >> 3;           // 0..31
    const int c4 = (tid & 7) << 2;     // 0..28
    v4f f = *(const v4f*)(W + (size_t)(k0 + r) * Ndim + n0 + c4);
#pragma unroll
    for (int e = 0; e < 4; ++e) tile[r][c4 + e] = f[e];
    __syncthreads();
#pragma unroll
    for (int e = 0; e < 4; ++e) {
        float val = tile[c4 + e][r];               // transposed read
        unsigned short h = f2bf_rn(val);
        unsigned short l = f2bf_rn(val - bf2f(h));
        size_t o = (size_t)(n0 + r) * Kdim + k0 + c4 + e;
        Th[o] = h;
        Tl[o] = l;
    }
}

// ---------------------------------------------------------------------------
// GEMM: C[M,N] = A[M,K] @ W[K,N] + bias[N].
//   A given as bf16 hi/lo row-major [M][K]; W given TRANSPOSED bf16 hi/lo
//   [N][K].  Block tile 128x128, BK=32, 8 waves (2x4), wave = 64x32 output.
//   Double-buffered LDS staged with global_load_async_to_lds_b128.
// ---------------------------------------------------------------------------
#define LDSTRIDE 40                 // shorts/row (80B): aligned, conflict-free b128
#define TILE_SHORTS (128 * LDSTRIDE)      // 5120 shorts = 10240 B per matrix
#define BUF_SHORTS  (4 * TILE_SHORTS)     // Ah, Al, Bh, Bl
#define GEMM_LDS_BYTES (2 * BUF_SHORTS * 2)   // 81920 B (double buffer)

__device__ __forceinline__ void stage_tiles(unsigned ldsbase,
                                            const unsigned short* Ah, const unsigned short* Al,
                                            const unsigned short* Bh, const unsigned short* Bl,
                                            int m0, int n0, int kt, int K, int tid) {
#pragma unroll
    for (int it = 0; it < 2; ++it) {
        int idx = tid + 256 * it;        // 0..511 -> 512 x 16B per matrix
        int row = idx >> 2;              // 0..127
        int kqb = (idx & 3) << 4;        // byte offset within 64B row: 0,16,32,48
        unsigned ldsA = ldsbase + (unsigned)(row * 80 + kqb);
        unsigned offA = (unsigned)(((m0 + row) * K + kt) * 2 + kqb);
        async_b128(ldsA,                     offA, Ah);
        async_b128(ldsA + 10240u,            offA, Al);
        unsigned ldsB = ldsbase + 20480u + (unsigned)(row * 80 + kqb);
        unsigned offB = (unsigned)(((n0 + row) * K + kt) * 2 + kqb);
        async_b128(ldsB,                     offB, Bh);
        async_b128(ldsB + 10240u,            offB, Bl);
    }
}

__global__ __launch_bounds__(256)
void gemm_bf16x3_async_kernel(const unsigned short* __restrict__ Ah,
                              const unsigned short* __restrict__ Al,
                              const unsigned short* __restrict__ BTh,
                              const unsigned short* __restrict__ BTl,
                              const float* __restrict__ bias, float* __restrict__ C,
                              int M, int N, int K) {
    extern __shared__ __align__(16) unsigned short smem[];   // 2 x 40960 B

    const int tid   = threadIdx.x;
    const int m0    = blockIdx.y * 128;
    const int n0    = blockIdx.x * 128;
    const int lane  = tid & 31;
    const int wave  = tid >> 5;
    const int half  = lane >> 4;
    const int lr    = lane & 15;
    const int waveM = wave & 1;    // 64-row band
    const int waveN = wave >> 1;   // 32-col band

    const unsigned lds0 = lds_addr_of(smem);

    v8f acc[4][2];
#pragma unroll
    for (int mi = 0; mi < 4; ++mi)
#pragma unroll
        for (int ni = 0; ni < 2; ++ni)
#pragma unroll
            for (int r = 0; r < 8; ++r) acc[mi][ni][r] = 0.0f;

    const int nsteps = K / 32;
    stage_tiles(lds0, Ah, Al, BTh, BTl, m0, n0, 0, K, tid);

    for (int s = 0; s < nsteps; ++s) {
        wait_async0();          // my async writes for buf[s&1] have landed
        __syncthreads();        // everyone's landed; everyone done with buf[(s+1)&1]
        if (s + 1 < nsteps)
            stage_tiles(lds0 + ((unsigned)((s + 1) & 1)) * 40960u,
                        Ah, Al, BTh, BTl, m0, n0, (s + 1) * 32, K, tid);

        const unsigned short* sA_hi = smem + (s & 1) * BUF_SHORTS;
        const unsigned short* sA_lo = sA_hi + TILE_SHORTS;
        const unsigned short* sB_hi = sA_hi + 2 * TILE_SHORTS;
        const unsigned short* sB_lo = sA_hi + 3 * TILE_SHORTS;

        // B fragments (contiguous k, same as A: lane<16 -> col=lr K0..15; lane>=16 -> col=lr K16..31)
        Frag bH[2], bL[2];
#pragma unroll
        for (int ni = 0; ni < 2; ++ni) {
            int col = waveN * 32 + ni * 16 + lr;
            int o = col * LDSTRIDE + half * 16;
            bH[ni].q[0] = *(const u32x4*)&sB_hi[o];
            bH[ni].q[1] = *(const u32x4*)&sB_hi[o + 8];
            bL[ni].q[0] = *(const u32x4*)&sB_lo[o];
            bL[ni].q[1] = *(const u32x4*)&sB_lo[o + 8];
        }
        // A fragments + 3x-bf16 WMMA accumulate
#pragma unroll
        for (int mi = 0; mi < 4; ++mi) {
            int row = waveM * 64 + mi * 16 + lr;
            // lane<16 holds K0..15; lane>=16 holds K16..31 (matches B)
            int o1 = row * LDSTRIDE + half * 16;
            int o2 = row * LDSTRIDE + half * 16 + 8;
            Frag aH, aL;
            aH.q[0] = *(const u32x4*)&sA_hi[o1];
            aH.q[1] = *(const u32x4*)&sA_hi[o2];
            aL.q[0] = *(const u32x4*)&sA_lo[o1];
            aL.q[1] = *(const u32x4*)&sA_lo[o2];
#pragma unroll
            for (int ni = 0; ni < 2; ++ni) {
                acc[mi][ni] = __builtin_amdgcn_wmma_f32_16x16x32_bf16(
                    false, aH.bf, false, bH[ni].bf, (short)0, acc[mi][ni], false, false);
                acc[mi][ni] = __builtin_amdgcn_wmma_f32_16x16x32_bf16(
                    false, aH.bf, false, bL[ni].bf, (short)0, acc[mi][ni], false, false);
                acc[mi][ni] = __builtin_amdgcn_wmma_f32_16x16x32_bf16(
                    false, aL.bf, false, bH[ni].bf, (short)0, acc[mi][ni], false, false);
            }
        }
        __syncthreads();
    }

    // epilogue: bias + store (C/D layout: row = r + 8*half, col = lr)
#pragma unroll
    for (int ni = 0; ni < 2; ++ni) {
        int col = n0 + waveN * 32 + ni * 16 + lr;
        float bv = bias[col];
#pragma unroll
        for (int mi = 0; mi < 4; ++mi) {
            int rowbase = m0 + waveM * 64 + mi * 16 + half * 8;
#pragma unroll
            for (int r = 0; r < 8; ++r)
                C[(size_t)(rowbase + r) * N + col] = acc[mi][ni][r] + bv;
        }
    }
}

// ---------------------------------------------------------------------------
// Filter MLP: h[k,d] for k < KTRUNC.  pos_emb -> gelu MLP -> * exp(-0.01 k).
// ---------------------------------------------------------------------------
__device__ __forceinline__ float gelu_tanh(float x) {
    float inner = 0.7978845608028654f * (x + 0.044715f * x * x * x);
    return 0.5f * x * (1.0f + tanhf(inner));
}

__global__ __launch_bounds__(256)
void filter_kernel(const float* __restrict__ W1, const float* __restrict__ b1,
                   const float* __restrict__ W2, const float* __restrict__ b2,
                   const float* __restrict__ W3, const float* __restrict__ b3,
                   float* __restrict__ hf) {
    __shared__ float pe[64], h1[64], h2[64];
    const int tid = threadIdx.x;
    const float t = (float)blockIdx.x;

    if (tid < 32) {
        float freq = expf((float)(2 * tid) * (-9.210340371976184f / 64.0f));
        float a = t * freq;
        pe[2 * tid]     = sinf(a);
        pe[2 * tid + 1] = cosf(a);
    }
    __syncthreads();
    if (tid < 64) {
        float s = b1[tid];
#pragma unroll 8
        for (int i = 0; i < 64; ++i) s += pe[i] * W1[i * 64 + tid];
        h1[tid] = gelu_tanh(s);
    }
    __syncthreads();
    if (tid < 64) {
        float s = b2[tid];
#pragma unroll 8
        for (int i = 0; i < 64; ++i) s += h1[i] * W2[i * 64 + tid];
        h2[tid] = gelu_tanh(s);
    }
    __syncthreads();
    const float decay = expf(-0.01f * t);
#pragma unroll
    for (int r = 0; r < 4; ++r) {
        int d = tid + 256 * r;
        float s = b3[d];
#pragma unroll 8
        for (int i = 0; i < 64; ++i) s += h2[i] * W3[(size_t)i * DD + d];
        hf[(size_t)blockIdx.x * DD + d] = s * decay;
    }
}

// ---------------------------------------------------------------------------
// Truncated causal depthwise conv + gating, in place over u:
//   u[b,t,d] <- u[b,t,d] * sum_{k<KTRUNC} h[k,d] * v[b,t-k,d]
// ---------------------------------------------------------------------------
#define CT 128
#define CK 128

__global__ __launch_bounds__(256)
void conv_gate_kernel(const float* __restrict__ v, const float* __restrict__ hf,
                      float* __restrict__ uz) {
    __shared__ float vbuf[(CT + CK) * 17];
    __shared__ float hbuf[CK * 16];

    const int tid = threadIdx.x;
    const int d0  = blockIdx.x * 16;
    const int t0  = blockIdx.y * CT;
    const int b   = blockIdx.z;
    const int d   = tid & 15;
    const int tg  = tid >> 4;
    const int tt0 = tg * 8;
    const size_t basebd = (size_t)b * LL * DD;

    float acc[8];
#pragma unroll
    for (int j = 0; j < 8; ++j) acc[j] = 0.0f;

    for (int kc = 0; kc < KTRUNC; kc += CK) {
#pragma unroll
        for (int it = 0; it < 16; ++it) {
            int i = tg + 16 * it;
            int tglob = t0 - kc - (CK - 1) + i;
            float val = 0.0f;
            if (tglob >= 0) val = v[basebd + (size_t)tglob * DD + d0 + d];
            vbuf[i * 17 + d] = val;
        }
#pragma unroll
        for (int it = 0; it < 8; ++it) {
            int kk = tg + 16 * it;
            hbuf[kk * 16 + d] = hf[(size_t)(kc + kk) * DD + d0 + d];
        }
        __syncthreads();

        float win[8];
#pragma unroll
        for (int j = 0; j < 8; ++j) win[j] = vbuf[(tt0 + 127 + j) * 17 + d];

#pragma unroll 4
        for (int k = 0; k < CK; ++k) {
            float hv = hbuf[k * 16 + d];
#pragma unroll
            for (int j = 0; j < 8; ++j) acc[j] += hv * win[j];
            if (k < CK - 1) {
#pragma unroll
                for (int j = 7; j > 0; --j) win[j] = win[j - 1];
                win[0] = vbuf[(tt0 + 126 - k) * 17 + d];
            }
        }
        __syncthreads();
    }

#pragma unroll
    for (int j = 0; j < 8; ++j) {
        size_t idx = basebd + (size_t)(t0 + tt0 + j) * DD + d0 + d;
        uz[idx] = uz[idx] * acc[j];
    }
}

// ---------------------------------------------------------------------------
extern "C" void kernel_launch(void* const* d_in, const int* in_sizes, int n_in,
                              void* d_out, int out_size, void* d_ws, size_t ws_size,
                              hipStream_t stream) {
    (void)in_sizes; (void)n_in; (void)out_size; (void)ws_size;
    const float* x  = (const float*)d_in[0];
    const float* Wu = (const float*)d_in[1];
    const float* bu = (const float*)d_in[2];
    const float* Wv = (const float*)d_in[3];
    const float* bv = (const float*)d_in[4];
    const float* W1 = (const float*)d_in[5];
    const float* b1 = (const float*)d_in[6];
    const float* W2 = (const float*)d_in[7];
    const float* b2 = (const float*)d_in[8];
    const float* W3 = (const float*)d_in[9];
    const float* b3 = (const float*)d_in[10];
    const float* Wo = (const float*)d_in[11];
    const float* bo = (const float*)d_in[12];
    float* out = (float*)d_out;

    const size_t nmat = (size_t)MM * DD;            // 16M elements
    float* u  = (float*)d_ws;                       // 64 MB (later holds z)
    float* v  = u + nmat;                           // 64 MB
    float* hf = v + nmat;                           // filter (pad to 2M floats)
    unsigned short* xh = (unsigned short*)(hf + 2 * 1024 * 1024);  // 32 MB (later zh)
    unsigned short* xl = xh + nmat;                               // 32 MB (later zl)
    unsigned short* WuTh = xl + nmat;                 // 6 x 2 MB weight splits
    unsigned short* WuTl = WuTh + (size_t)DD * DD;
    unsigned short* WvTh = WuTl + (size_t)DD * DD;
    unsigned short* WvTl = WvTh + (size_t)DD * DD;
    unsigned short* WoTh = WvTl + (size_t)DD * DD;
    unsigned short* WoTl = WoTh + (size_t)DD * DD;

    const int n4 = (int)(nmat / 4);
    dim3 gT(DD / 32, DD / 32);
    dim3 gg(DD / 128, MM / 128);

    // pre-split operands to bf16 hi/lo (weights also transposed)
    split_bf16_kernel<<<(n4 + 255) / 256, 256, 0, stream>>>(x, xh, xl, n4);
    splitT_bf16_kernel<<<gT, 256, 0, stream>>>(Wu, WuTh, WuTl, DD, DD);
    splitT_bf16_kernel<<<gT, 256, 0, stream>>>(Wv, WvTh, WvTl, DD, DD);
    splitT_bf16_kernel<<<gT, 256, 0, stream>>>(Wo, WoTh, WoTl, DD, DD);

    gemm_bf16x3_async_kernel<<<gg, 256, GEMM_LDS_BYTES, stream>>>(
        xh, xl, WuTh, WuTl, bu, u, MM, DD, DD);
    gemm_bf16x3_async_kernel<<<gg, 256, GEMM_LDS_BYTES, stream>>>(
        xh, xl, WvTh, WvTl, bv, v, MM, DD, DD);

    filter_kernel<<<KTRUNC, 256, 0, stream>>>(W1, b1, W2, b2, W3, b3, hf);
    conv_gate_kernel<<<dim3(DD / 16, LL / CT, BB), 256, 0, stream>>>(v, hf, u);

    // z = u (in place); re-split for final GEMM, reusing xh/xl space
    split_bf16_kernel<<<(n4 + 255) / 256, 256, 0, stream>>>(u, xh, xl, n4);
    gemm_bf16x3_async_kernel<<<gg, 256, GEMM_LDS_BYTES, stream>>>(
        xh, xl, WoTh, WoTl, bo, out, MM, DD, DD);
}
